// TrajPredictor_21002390077960
// MI455X (gfx1250) — compile-verified, matrix operand order checked by reading, not run on hardware
//
#include <hip/hip_runtime.h>

#define B_  16
#define S_  64
#define V_  4000
#define E_  128
#define H_  256
#define FH_ 1024   // 4*H
#define TE_ 384    // 3*E (== E+H for W1 too)

typedef __attribute__((ext_vector_type(16))) _Float16 v16h;
typedef __attribute__((ext_vector_type(8)))  _Float16 v8h;
typedef __attribute__((ext_vector_type(8)))  float    v8f;

// ---------------- WMMA helpers (16x16x32 f16 -> f32) ----------------

__device__ __forceinline__ v8f wmma_f16(v16h a, v16h b, v8f c) {
  return __builtin_amdgcn_wmma_f32_16x16x32_f16(false, a, false, b, (short)0, c,
                                                false, false);
}

// A/B fragment load from a row-major [16 x K] tile; lanerow = base of this
// lane's row (row index = lane&15 chosen by caller), kbase = K offset (mult 32).
// Layout per ISA 7.12.2: lane<16 holds K {kbase..+7, kbase+16..+23},
// lane>=16 holds K {kbase+8..+15, kbase+24..+31}.
__device__ __forceinline__ v16h frag_ld_ptr(const _Float16* lanerow, int kbase) {
  int lane = threadIdx.x & 31;
  const _Float16* p = lanerow + kbase + ((lane >> 4) << 3);
  v8h lo = *(const v8h*)p;
  v8h hi = *(const v8h*)(p + 16);
  v16h r;
#pragma unroll
  for (int i = 0; i < 8; ++i) { r[i] = lo[i]; r[i + 8] = hi[i]; }
  return r;
}

// C/D store: element r -> (M = moff+r, N = lane&15)
__device__ __forceinline__ void frag_st(float* base, int ldc, v8f c) {
  int lane = threadIdx.x & 31;
  int lm = lane & 15, moff = (lane >> 4) << 3;
#pragma unroll
  for (int r = 0; r < 8; ++r) base[(size_t)(moff + r) * ldc + lm] = c[r];
}

__device__ __forceinline__ float sigm(float x) {
  return 1.0f / (1.0f + __expf(-x));
}

// ---------------- conversion / prep kernels ----------------

__global__ void k_cvt(const float* __restrict__ src, int src_rows, int src_cols,
                      int src_ld, int col0, _Float16* __restrict__ dst,
                      int dst_rows, int dst_cols) {
  int idx = blockIdx.x * blockDim.x + threadIdx.x;
  if (idx >= dst_rows * dst_cols) return;
  int r = idx / dst_cols, c = idx - r * dst_cols;
  float v = 0.0f;
  if (r < src_rows && c < src_cols) v = src[(size_t)r * src_ld + col0 + c];
  dst[idx] = (_Float16)v;
}

__global__ void k_bias(const float* __restrict__ b_ih, const float* __restrict__ b_hh,
                       const float* __restrict__ b1, float* __restrict__ biasg,
                       float* __restrict__ b1pad) {
  int i = blockIdx.x * blockDim.x + threadIdx.x;
  if (i < FH_) biasg[i] = b_ih[i] + b_hh[i];
  if (i < 32) b1pad[i] = (i < 30) ? b1[i] : 0.0f;
}

// ---------------- x-contribution: emb[seq] @ W_ih[:, :E].T  [1024 x 1024] ----

__global__ void k_xcontrib(const int* __restrict__ seq,
                           const _Float16* __restrict__ emb16,
                           const _Float16* __restrict__ wihA,
                           float* __restrict__ xc) {
  int wave = (blockIdx.x * blockDim.x + threadIdx.x) >> 5;  // 4096 waves
  int mt = wave >> 6;   // 64 row tiles over B*S
  int nt = wave & 63;   // 64 col tiles over 4H
  int lane = threadIdx.x & 31;
  int lm = lane & 15;
  int token = seq[mt * 16 + lm];                       // fused embedding gather
  const _Float16* Arow = emb16 + (size_t)token * E_;
  const _Float16* Brow = wihA + ((size_t)nt * 16 + lm) * E_;
  v8f acc = {};
#pragma unroll
  for (int kk = 0; kk < 4; ++kk)
    acc = wmma_f16(frag_ld_ptr(Arow, kk * 32), frag_ld_ptr(Brow, kk * 32), acc);
  frag_st(xc + (size_t)mt * 16 * FH_ + nt * 16, FH_, acc);
}

// ---------------- sd-contribution: sd_emb @ W_ih[:, E:].T + (b_ih+b_hh) -------

__global__ void __launch_bounds__(256)
k_sdcontrib(const int* __restrict__ seq_sd, const _Float16* __restrict__ emb16,
            const _Float16* __restrict__ wihSD, const float* __restrict__ biasg,
            float* __restrict__ sdc) {
  __shared__ __align__(16) _Float16 sdA[16 * 256];
  int tid = threadIdx.x;
  for (int i = tid; i < 16 * 256; i += 256) {
    int b = i >> 8, k = i & 255;
    int token = seq_sd[b * 2 + (k >> 7)];
    sdA[i] = emb16[(size_t)token * E_ + (k & 127)];
  }
  __syncthreads();
  int wave = tid >> 5, lane = tid & 31, lm = lane & 15;
  const _Float16* Arow = &sdA[lm * 256];
  for (int nt = wave * 8; nt < wave * 8 + 8; ++nt) {   // 8 waves x 8 tiles = 64
    const _Float16* Brow = wihSD + ((size_t)nt * 16 + lm) * 256;
    v8f acc = {};
#pragma unroll
    for (int kk = 0; kk < 8; ++kk)
      acc = wmma_f16(frag_ld_ptr(Arow, kk * 32), frag_ld_ptr(Brow, kk * 32), acc);
    float bb = biasg[nt * 16 + lm];
#pragma unroll
    for (int r = 0; r < 8; ++r) acc[r] += bb;
    frag_st(sdc + nt * 16, FH_, acc);
  }
}

// ---------------- persistent LSTM: 64 recurrent steps, 1 workgroup ----------
// Wave w owns hidden cols w*16..w*16+15 for ALL four gates (n-tiles
// w, w+16, w+32, w+48) so i/f/g/o nonlinearity completes in-wave; cell state c
// lives in the accumulator register layout. h (f16) shared via 8KB LDS.

__global__ void __launch_bounds__(512)
k_lstm(const float* __restrict__ xc, const float* __restrict__ sdc,
       const _Float16* __restrict__ whh16, const int* __restrict__ seqlen,
       _Float16* __restrict__ out16) {
  __shared__ __align__(16) _Float16 h16[16 * 256];
  int tid = threadIdx.x;
  int wave = tid >> 5, lane = tid & 31;
  int lm = lane & 15, moff = (lane >> 4) << 3;
  int hc = wave * 16 + lm;  // hidden column this lane produces

  const _Float16* Br[4];
#pragma unroll
  for (int g = 0; g < 4; ++g)
    Br[g] = whh16 + ((size_t)((g * 16 + wave) * 16 + lm)) * H_;

  float sdp[4][8];
#pragma unroll
  for (int g = 0; g < 4; ++g)
#pragma unroll
    for (int r = 0; r < 8; ++r)
      sdp[g][r] = sdc[(size_t)(moff + r) * FH_ + g * H_ + hc];

  int slen[8];
#pragma unroll
  for (int r = 0; r < 8; ++r) slen[r] = seqlen[moff + r];

  for (int i = tid; i < 16 * 256; i += 512) h16[i] = (_Float16)0.0f;
  v8f cacc = {};
  __syncthreads();

  for (int t = 0; t < S_; ++t) {
    // read all of h before anyone overwrites it
    v16h afr[8];
#pragma unroll
    for (int kk = 0; kk < 8; ++kk) afr[kk] = frag_ld_ptr(&h16[lm * 256], kk * 32);
    __syncthreads();

    v8f acc[4];
#pragma unroll
    for (int g = 0; g < 4; ++g)
#pragma unroll
      for (int r = 0; r < 8; ++r)
        acc[g][r] = xc[(size_t)((moff + r) * S_ + t) * FH_ + g * H_ + hc] + sdp[g][r];

#pragma unroll
    for (int kk = 0; kk < 8; ++kk) {
      v16h a = afr[kk];
#pragma unroll
      for (int g = 0; g < 4; ++g)
        acc[g] = wmma_f16(a, frag_ld_ptr(Br[g], kk * 32), acc[g]);  // W_hh from L2
    }

#pragma unroll
    for (int r = 0; r < 8; ++r) {
      int b = moff + r;
      float si = sigm(acc[0][r]);
      float sf = sigm(acc[1][r]);
      float gg = tanhf(acc[2][r]);
      float so = sigm(acc[3][r]);
      float ct = sf * cacc[r] + si * gg;
      cacc[r] = ct;
      float hv = so * tanhf(ct);
      h16[b * 256 + hc] = (_Float16)hv;
      out16[(size_t)(b * S_ + t) * H_ + hc] =
          (t < slen[r]) ? (_Float16)hv : (_Float16)0.0f;  // pad_packed mask
    }
    __syncthreads();
  }
}

// ---------------- generic WMMA GEMM: C[Mt*16 x Nt*16] = A @ Bt.T (+bias) ----

__global__ void k_gemm(const _Float16* __restrict__ A, int lda,
                       const _Float16* __restrict__ Bt, int ldb,
                       const float* __restrict__ bias, float* __restrict__ C,
                       int ldc, int nWaves, int nTiles, int kSteps) {
  int wv = (blockIdx.x * blockDim.x + threadIdx.x) >> 5;
  if (wv >= nWaves) return;  // wave-uniform exit keeps EXEC all-ones for WMMA
  int mt = wv / nTiles, nt = wv - mt * nTiles;
  int lane = threadIdx.x & 31, lm = lane & 15;
  const _Float16* Arow = A + ((size_t)mt * 16 + lm) * lda;
  const _Float16* Brow = Bt + ((size_t)nt * 16 + lm) * ldb;
  v8f acc = {};
  for (int kk = 0; kk < kSteps; ++kk)
    acc = wmma_f16(frag_ld_ptr(Arow, kk * 32), frag_ld_ptr(Brow, kk * 32), acc);
  if (bias) {
    float bb = bias[nt * 16 + lm];
#pragma unroll
    for (int r = 0; r < 8; ++r) acc[r] += bb;
  }
  frag_st(C + (size_t)mt * 16 * ldc + nt * 16, ldc, acc);
}

// ---------------- final fused relu-dot: logits[bs,v] ------------------------
// bs-tiled: each thread keeps poi[v,0..29] in registers and reuses it across
// 16 bs rows (out_part rows live in LDS) -> 16x less L2 read traffic.

#define BS_TILE 16

__global__ void __launch_bounds__(256)
k_final(const float* __restrict__ poi, const float* __restrict__ outp,
        const float* __restrict__ W2, const float* __restrict__ b2,
        float* __restrict__ logits) {
  __shared__ float op[BS_TILE][32];
  __shared__ float w2s[32];
  int tid = threadIdx.x;
  int bs0 = blockIdx.y * BS_TILE;
  for (int i = tid; i < BS_TILE * 32; i += 256)
    op[i >> 5][i & 31] = outp[(size_t)(bs0 + (i >> 5)) * 32 + (i & 31)];
  if (tid < 32) w2s[tid] = (tid < 30) ? W2[tid] : 0.0f;
  __syncthreads();

  int v = blockIdx.x * 256 + tid;
  if (v < V_) {
    float pr[30];
#pragma unroll
    for (int j = 0; j < 30; ++j) pr[j] = poi[(size_t)v * 32 + j];
    float bias2 = b2[0];
#pragma unroll 4
    for (int b = 0; b < BS_TILE; ++b) {
      float s = bias2;
#pragma unroll
      for (int j = 0; j < 30; ++j)
        s = fmaf(fmaxf(pr[j] + op[b][j], 0.0f), w2s[j], s);
      logits[(size_t)(bs0 + b) * V_ + v] = s;
    }
  }
}

// ---------------- launcher ----------------

extern "C" void kernel_launch(void* const* d_in, const int* in_sizes, int n_in,
                              void* d_out, int out_size, void* d_ws, size_t ws_size,
                              hipStream_t stream) {
  const int*   seq   = (const int*)d_in[0];
  const int*   seqsd = (const int*)d_in[1];
  const int*   slen  = (const int*)d_in[2];
  const float* emb   = (const float*)d_in[3];
  const float* Wih   = (const float*)d_in[4];
  const float* Whh   = (const float*)d_in[5];
  const float* bih   = (const float*)d_in[6];
  const float* bhh   = (const float*)d_in[7];
  const float* W1    = (const float*)d_in[8];
  const float* b1    = (const float*)d_in[9];
  const float* W2    = (const float*)d_in[10];
  const float* b2    = (const float*)d_in[11];
  float* logits = (float*)d_out;

  char* ws = (char*)d_ws;
  size_t off = 0;
  auto take = [&](size_t bytes) -> char* {
    char* p = ws + off;
    off += (bytes + 255) & ~(size_t)255;
    return p;
  };
  _Float16* emb16 = (_Float16*)take((size_t)V_ * E_ * 2);
  _Float16* wihA  = (_Float16*)take((size_t)FH_ * E_ * 2);
  _Float16* wihSD = (_Float16*)take((size_t)FH_ * 2 * E_ * 2);
  _Float16* whh16 = (_Float16*)take((size_t)FH_ * H_ * 2);
  _Float16* w1a   = (_Float16*)take((size_t)32 * E_ * 2);
  _Float16* w1b   = (_Float16*)take((size_t)32 * H_ * 2);
  float* biasg = (float*)take((size_t)FH_ * 4);
  float* b1pad = (float*)take(32 * 4);
  float* xc    = (float*)take((size_t)B_ * S_ * FH_ * 4);
  float* sdc   = (float*)take((size_t)B_ * FH_ * 4);
  _Float16* out16 = (_Float16*)take((size_t)B_ * S_ * H_ * 2);
  float* outp = (float*)take((size_t)B_ * S_ * 32 * 4);
  float* poi  = (float*)take((size_t)V_ * 32 * 4);

  auto cdiv = [](int a, int b) { return (a + b - 1) / b; };

  // f32 -> f16 weight/embedding prep
  k_cvt<<<cdiv(V_ * E_, 256), 256, 0, stream>>>(emb, V_, E_, E_, 0, emb16, V_, E_);
  k_cvt<<<cdiv(FH_ * E_, 256), 256, 0, stream>>>(Wih, FH_, E_, TE_, 0, wihA, FH_, E_);
  k_cvt<<<cdiv(FH_ * 2 * E_, 256), 256, 0, stream>>>(Wih, FH_, 2 * E_, TE_, E_, wihSD, FH_, 2 * E_);
  k_cvt<<<cdiv(FH_ * H_, 256), 256, 0, stream>>>(Whh, FH_, H_, H_, 0, whh16, FH_, H_);
  k_cvt<<<cdiv(32 * E_, 256), 256, 0, stream>>>(W1, 30, E_, TE_, 0, w1a, 32, E_);
  k_cvt<<<cdiv(32 * H_, 256), 256, 0, stream>>>(W1, 30, H_, TE_, E_, w1b, 32, H_);
  k_bias<<<4, 256, 0, stream>>>(bih, bhh, b1, biasg, b1pad);

  // batched input contribution (WMMA, fused embedding gather): 4096 waves
  k_xcontrib<<<512, 256, 0, stream>>>(seq, emb16, wihA, xc);
  // per-batch constant contribution + bias: 1 block
  k_sdcontrib<<<1, 256, 0, stream>>>(seqsd, emb16, wihSD, biasg, sdc);
  // recurrent LSTM: 1 persistent workgroup, 16 waves
  k_lstm<<<1, 512, 0, stream>>>(xc, sdc, whh16, slen, out16);

  // out_part = lstm_out @ W1[:,E:].T + b1  : M=1024, N=32, K=256
  k_gemm<<<16, 256, 0, stream>>>(out16, H_, w1b, H_, b1pad, outp, 32, 128, 2, 8);
  // poi_part = emb @ W1[:,:E].T            : M=4000, N=32, K=128
  k_gemm<<<63, 256, 0, stream>>>(emb16, E_, w1a, E_, nullptr, poi, 32, 500, 2, 4);

  // logits = relu(poi + out_part) . W2 + b2 (bs-tiled for poi reuse)
  k_final<<<dim3(16, (B_ * S_) / BS_TILE), 256, 0, stream>>>(poi, outp, W2, b2, logits);
}
